// GraphAttention_30958124269885
// MI455X (gfx1250) — matrix operand order repaired and checked by reading, not compiled, not run on hardware
//
#include <hip/hip_runtime.h>
#include <math.h>

typedef __attribute__((ext_vector_type(16))) _Float16 v16h;
typedef __attribute__((ext_vector_type(8)))  float    v8f;

#define NN      8192
#define IN_DIM  256
#define OUT_DIM 64
#define SLOPE   0.2f

// ---------------------------------------------------------------------------
// Kernel 0: Wh = h @ W  (fp32), f_i = Wh@a[:64], f_j = Wh@a[64:],
//           and WhT f16 [OUT_DIM][NN] (transposed, for WMMA B operand loads).
// ---------------------------------------------------------------------------
__global__ __launch_bounds__(256) void k_proj(const float* __restrict__ h,
                                              const float* __restrict__ W,
                                              const float* __restrict__ a,
                                              _Float16* __restrict__ WhT,
                                              float* __restrict__ fi_arr,
                                              float* __restrict__ fj_arr) {
    __shared__ float s_h[4][IN_DIM];
    __shared__ float s_red[2][4][64];
    const int t   = threadIdx.x;
    const int sub = t >> 6;        // 0..3 (row within group of 4)
    const int col = t & 63;        // output column
    const int rowBase = blockIdx.x * 32;

    for (int it = 0; it < 8; ++it) {
        __syncthreads();
        for (int q = 0; q < 4; ++q) {
            const int idx = q * 256 + t;
            s_h[idx >> 8][idx & 255] =
                h[(rowBase + it * 4 + (idx >> 8)) * IN_DIM + (idx & 255)];
        }
        __syncthreads();

        const int row = rowBase + it * 4 + sub;
        float acc = 0.f;
        #pragma unroll 8
        for (int k = 0; k < IN_DIM; ++k)
            acc = fmaf(s_h[sub][k], W[k * OUT_DIM + col], acc);

        WhT[col * NN + row] = (_Float16)acc;

        s_red[0][sub][col] = acc * a[col];
        s_red[1][sub][col] = acc * a[OUT_DIM + col];
        __syncthreads();
        for (int off = 32; off > 0; off >>= 1) {
            if (col < off) {
                s_red[0][sub][col] += s_red[0][sub][col + off];
                s_red[1][sub][col] += s_red[1][sub][col + off];
            }
            __syncthreads();
        }
        if (col == 0) {
            fi_arr[row] = s_red[0][sub][0];
            fj_arr[row] = s_red[1][sub][0];
        }
    }
}

// ---------------------------------------------------------------------------
// Kernel 1: one block per row i. Read adj[i,:] ONCE, ballot-pack to bitmask,
// compute row max m_i and denom l_i. Fully branchless inner loops.
// ---------------------------------------------------------------------------
__global__ __launch_bounds__(256) void k_rowstats(const int* __restrict__ adj,
                                                  const float* __restrict__ fi_arr,
                                                  const float* __restrict__ fj_arr,
                                                  unsigned* __restrict__ bits,
                                                  float* __restrict__ m_arr,
                                                  float* __restrict__ l_arr) {
    __shared__ float s_fj[NN];
    __shared__ float s_red[256];
    const int i = blockIdx.x;
    const int t = threadIdx.x;
    for (int q = 0; q < NN / 256; ++q) s_fj[q * 256 + t] = fj_arr[q * 256 + t];
    __syncthreads();

    const float fi = fi_arr[i];
    const int   w  = t >> 5;          // wave id within block
    unsigned myBits = 0;
    float mloc = -INFINITY;

    #pragma unroll 4
    for (int it = 0; it < 32; ++it) {
        const int j   = it * 256 + t;
        const bool pr = adj[(long)i * NN + j] > 0;
        const unsigned word = __builtin_amdgcn_ballot_w32(pr);
        if ((t & 31) == 0) bits[i * 256 + it * 8 + w] = word;
        float e = fi + s_fj[j];
        e = e >= 0.f ? e : SLOPE * e;
        mloc = fmaxf(mloc, pr ? e : -INFINITY);
        myBits |= ((unsigned)pr) << it;
    }

    s_red[t] = mloc;
    __syncthreads();
    for (int off = 128; off > 0; off >>= 1) {
        if (t < off) s_red[t] = fmaxf(s_red[t], s_red[t + off]);
        __syncthreads();
    }
    const float m = s_red[0];
    __syncthreads();

    float sloc = 0.f;
    #pragma unroll 4
    for (int it = 0; it < 32; ++it) {
        const int j = it * 256 + t;
        float e = fi + s_fj[j];
        e = e >= 0.f ? e : SLOPE * e;
        const float msk = (float)((myBits >> it) & 1u);
        sloc += __expf(fminf(e - m, 0.f)) * msk;   // clamp avoids inf*0
    }
    s_red[t] = sloc;
    __syncthreads();
    for (int off = 128; off > 0; off >>= 1) {
        if (t < off) s_red[t] += s_red[t + off];
        __syncthreads();
    }
    if (t == 0) { m_arr[i] = m; l_arr[i] = s_red[0]; }
}

// ---------------------------------------------------------------------------
// Kernel 2: fused softmax * Wh via v_wmma_f32_16x16x32_f16.
// One wave per 16-row tile; 4 chunks (128 columns) per outer iteration.
// A built branchlessly in the 16-bit A layout; f_j read as float4 from LDS.
// ---------------------------------------------------------------------------
__global__ __launch_bounds__(256) void k_attn(const unsigned* __restrict__ bits,
                                              const float* __restrict__ fi_arr,
                                              const float* __restrict__ fj_arr,
                                              const float* __restrict__ m_arr,
                                              const float* __restrict__ l_arr,
                                              const _Float16* __restrict__ WhT,
                                              float* __restrict__ out) {
    __shared__ float s_fj[NN];
    const int t = threadIdx.x;
    for (int q = 0; q < NN / 256; ++q) s_fj[q * 256 + t] = fj_arr[q * 256 + t];
    __syncthreads();

    const int wave = t >> 5;
    const int lane = t & 31;
    const int row0 = (blockIdx.x * 8 + wave) * 16;

    const int   rA   = row0 + (lane & 15);           // A-operand row for this lane
    const float fi   = fi_arr[rA];
    const float m    = m_arr[rA];
    const float l    = l_arr[rA];
    const float invl = l > 0.f ? 1.f / l : 0.f;
    const int   half = lane >> 4;                    // 0 or 1
    const int   kofs = half * 8;                     // A layout K offset
    const int   bofs = half * 16;                    // B layout K offset
    const int   ncol = lane & 15;                    // B layout N = lane%16

    v8f acc0 = {}, acc1 = {}, acc2 = {}, acc3 = {};

    for (int c = 0; c < NN / 32; c += 4) {
        const uint4 wv = *(const uint4*)&bits[rA * 256 + c];

        #pragma unroll
        for (int s = 0; s < 4; ++s) {
            const unsigned w = ((const unsigned*)&wv)[s];
            const int jb = (c + s) * 32;

            // 16 f_j values per lane: one aligned 64B run -> 4x ds_load_b128
            const float4 f0 = *(const float4*)&s_fj[jb + bofs];
            const float4 f1 = *(const float4*)&s_fj[jb + bofs + 4];
            const float4 f2 = *(const float4*)&s_fj[jb + bofs + 8];
            const float4 f3 = *(const float4*)&s_fj[jb + bofs + 12];
            const float fjv[16] = {f0.x, f0.y, f0.z, f0.w,
                                   f1.x, f1.y, f1.z, f1.w,
                                   f2.x, f2.y, f2.z, f2.w,
                                   f3.x, f3.y, f3.z, f3.w};

            // A (16x32 f16): lane<16 holds K in {0..15}, lane>=16 {16..31}
            v16h A;
            #pragma unroll
            for (int e = 0; e < 16; ++e) {
                const int k = e + bofs;
                float ev = fi + fjv[e];
                ev = ev >= 0.f ? ev : SLOPE * ev;
                const float msk = (float)((w >> k) & 1u);
                // clamp: valid entries have ev-m <= 0; prevents inf*0 for invalid
                const float p = __expf(fminf(ev - m, 0.f)) * invl * msk;
                A[e] = (_Float16)p;
            }

            // B (32x16 f16): lane = N, K half chosen by lane>=16 -> contiguous
            const v16h b0 = *(const v16h*)&WhT[( 0 + ncol) * NN + jb + bofs];
            const v16h b1 = *(const v16h*)&WhT[(16 + ncol) * NN + jb + bofs];
            const v16h b2 = *(const v16h*)&WhT[(32 + ncol) * NN + jb + bofs];
            const v16h b3 = *(const v16h*)&WhT[(48 + ncol) * NN + jb + bofs];

            acc0 = __builtin_amdgcn_wmma_f32_16x16x32_f16(false, A, false, b0, (short)0, acc0, false, false);
            acc1 = __builtin_amdgcn_wmma_f32_16x16x32_f16(false, A, false, b1, (short)0, acc1, false, false);
            acc2 = __builtin_amdgcn_wmma_f32_16x16x32_f16(false, A, false, b2, (short)0, acc2, false, false);
            acc3 = __builtin_amdgcn_wmma_f32_16x16x32_f16(false, A, false, b3, (short)0, acc3, false, false);
        }
    }

    // C/D layout: VGPR r -> M = r + 8*half, N = lane%16
    #pragma unroll
    for (int r = 0; r < 8; ++r) {
        const int row = row0 + r + half * 8;
        out[row * OUT_DIM +  0 + ncol] = acc0[r];
        out[row * OUT_DIM + 16 + ncol] = acc1[r];
        out[row * OUT_DIM + 32 + ncol] = acc2[r];
        out[row * OUT_DIM + 48 + ncol] = acc3[r];
    }
}

// ---------------------------------------------------------------------------
extern "C" void kernel_launch(void* const* d_in, const int* in_sizes, int n_in,
                              void* d_out, int out_size, void* d_ws, size_t ws_size,
                              hipStream_t stream) {
    const float* h   = (const float*)d_in[0];
    const int*   adj = (const int*)d_in[1];
    const float* W   = (const float*)d_in[2];
    const float* a   = (const float*)d_in[3];
    float*       out = (float*)d_out;

    // workspace carve-out (total ~9.2 MB)
    char* ws = (char*)d_ws;
    _Float16* WhT  = (_Float16*)ws;                                              // 1 MB
    unsigned* bits = (unsigned*)(ws + (size_t)OUT_DIM * NN * sizeof(_Float16));  // 8 MB
    float* fi = (float*)(ws + (size_t)OUT_DIM * NN * sizeof(_Float16)
                            + (size_t)NN * (NN / 32) * sizeof(unsigned));
    float* fj = fi + NN;
    float* mm = fj + NN;
    float* ll = mm + NN;

    k_proj    <<<NN / 32,       256, 0, stream>>>(h, W, a, WhT, fi, fj);
    k_rowstats<<<NN,            256, 0, stream>>>(adj, fi, fj, bits, mm, ll);
    k_attn    <<<NN / (16 * 8), 256, 0, stream>>>(bits, fi, fj, mm, ll, WhT, out);
}
